// MoEBlock_58969900974708
// MI455X (gfx1250) — matrix operand order repaired and checked by reading, not compiled, not run on hardware
//
#include <hip/hip_runtime.h>
#include <hip/hip_bf16.h>
#include <math.h>

// ---------------------------------------------------------------------------
// Types for CDNA5 WMMA (gfx1250, wave32)
// ---------------------------------------------------------------------------
typedef __bf16 bf16;
typedef __attribute__((ext_vector_type(4)))  float v4f;
typedef __attribute__((ext_vector_type(8)))  float v8f;
typedef __attribute__((ext_vector_type(8)))  bf16  v8bf;
typedef __attribute__((ext_vector_type(16))) bf16  v16bf;

#define WMMA_BF16(a, b, c) \
  __builtin_amdgcn_wmma_f32_16x16x32_bf16(false, (a), false, (b), (short)0, (c), false, false)

// Problem constants (reference: B=2,S=2048,D=1024,H=16,E=8,K=2,DFF=4096)
#define SEQ    2048
#define NBATCH 2
#define TOK    4096   // NBATCH*SEQ
#define DMODEL 1024
#define NH     16
#define HDIM   64
#define NEXP   8
#define DFF    4096

static __device__ __forceinline__ v16bf cat16(v8bf lo, v8bf hi) {
  v16bf r;
#pragma unroll
  for (int i = 0; i < 8; i++) { r[i] = lo[i]; r[i + 8] = hi[i]; }
  return r;
}

static __device__ __forceinline__ float gelu_tanh(float x) {
  return 0.5f * x * (1.0f + tanhf(0.7978845608028654f * (x + 0.044715f * x * x * x)));
}

// ---------------------------------------------------------------------------
// 16-lane xor "shuffle" for reductions. All xor masks are < 16, i.e. row-local
// on wave32, so DPP16 ROW_XMASK (ctrl = 0x160 | mask) does this in the VALU
// pipe with no LDS traffic (vs __shfl_xor -> ds_bpermute_b32).
// ---------------------------------------------------------------------------
#if __has_builtin(__builtin_amdgcn_update_dpp)
template <int MASK>
static __device__ __forceinline__ float xor16(float v) {
  return __int_as_float(
      __builtin_amdgcn_update_dpp(0, __float_as_int(v), 0x160 | MASK, 0xf, 0xf, true));
}
#else
template <int MASK>
static __device__ __forceinline__ float xor16(float v) { return __shfl_xor(v, MASK); }
#endif

static __device__ __forceinline__ float rowmax16(float v) {
  v = fmaxf(v, xor16<1>(v));
  v = fmaxf(v, xor16<2>(v));
  v = fmaxf(v, xor16<4>(v));
  v = fmaxf(v, xor16<8>(v));
  return v;
}
static __device__ __forceinline__ float rowsum16(float v) {
  v += xor16<1>(v);
  v += xor16<2>(v);
  v += xor16<4>(v);
  v += xor16<8>(v);
  return v;
}

// ---------------------------------------------------------------------------
// Generic tiled WMMA GEMM:  C[n,m] = sum_k A[n,k] * W[m,k]
//   A: bf16 row-major (lda = K stride), W: f32 row-major (ldb = K stride),
//   converted f32->bf16 while staging into LDS (weights are single-use).
// Tile 128x128x64, 256 threads = 8 waves, each wave -> 64x32 (4x2 WMMA tiles),
// 16 WMMAs per barrier round. Epilogue fuses GELU / row scale / residual add.
// ---------------------------------------------------------------------------
template <int ACT, bool ADD, bool SCALE, bool OUTBF>
__global__ __launch_bounds__(256) void gemm_xWT(
    const bf16* __restrict__ A, int lda,
    const float* __restrict__ W, int ldb,
    const float* __restrict__ Cin,
    float* __restrict__ outF, bf16* __restrict__ outB, int ldo,
    const float* __restrict__ rowscale, int rsStride,
    int Kdim) {
  __shared__ alignas(16) bf16 As[128][72];
  __shared__ alignas(16) bf16 Bs[128][72];

  const int tid  = threadIdx.x;
  const int lane = tid & 31;
  const int wid  = tid >> 5;
  const int wrow = (wid & 1) * 64;   // token offset of this wave inside tile
  const int wcol = (wid >> 1) * 32;  // output offset of this wave inside tile
  const int rowBase = blockIdx.y * 128;
  const int colBase = blockIdx.x * 128;
  const int l15  = lane & 15;
  const int aoff = (lane & 16) ? 8 : 0;   // A-operand K sub-offset per ISA layout
  const int boff = (lane & 16) ? 8 : 0;   // B-operand K sub-offset per ISA layout

  v8f acc[4][2] = {};

  for (int k0 = 0; k0 < Kdim; k0 += 64) {
    // ---- stage A (bf16 copy) and W (f32 -> bf16 convert) into LDS ----
#pragma unroll
    for (int it = 0; it < 4; it++) {
      int c  = tid + it * 256;          // 0..1023 chunks of 8 elements
      int r  = c >> 3;
      int cc = (c & 7) * 8;
      const bf16*  ap = &A[(size_t)(rowBase + r) * lda + k0 + cc];
      const float* wp = &W[(size_t)(colBase + r) * ldb + k0 + cc];
      *(v8bf*)&As[r][cc] = *(const v8bf*)ap;
      v4f f0 = *(const v4f*)wp;
      v4f f1 = *(const v4f*)(wp + 4);
      v8bf wb;
#pragma unroll
      for (int q = 0; q < 4; q++) { wb[q] = (bf16)f0[q]; wb[q + 4] = (bf16)f1[q]; }
      *(v8bf*)&Bs[r][cc] = wb;
      if (it == 0 && k0 + 64 < Kdim) {   // pull next K-tile toward cache
        __builtin_prefetch(ap + 64, 0, 3);
        __builtin_prefetch(wp + 64, 0, 3);
      }
    }
    __syncthreads();

    // ---- 16 WMMAs per round (fragments follow exact CDNA5 VGPR layouts) ----
#pragma unroll
    for (int kk = 0; kk < 64; kk += 32) {
      v16bf afr[4], bfr[2];
#pragma unroll
      for (int t = 0; t < 4; t++) {
        const bf16* p = &As[wrow + t * 16 + l15][kk + aoff];
        afr[t] = cat16(*(const v8bf*)p, *(const v8bf*)(p + 16));
      }
#pragma unroll
      for (int j = 0; j < 2; j++) {
        const bf16* p = &Bs[wcol + j * 16 + l15][kk + boff];
        bfr[j] = cat16(*(const v8bf*)p, *(const v8bf*)(p + 16));
      }
#pragma unroll
      for (int t = 0; t < 4; t++)
#pragma unroll
        for (int j = 0; j < 2; j++)
          acc[t][j] = WMMA_BF16(afr[t], bfr[j], acc[t][j]);
    }
    __syncthreads();
  }

  // ---- fused epilogue ----
  const int rlo = (lane >> 4) << 3;  // C/D layout: lanes 0-15 rows 0..7, 16-31 rows 8..15
#pragma unroll
  for (int t = 0; t < 4; t++)
#pragma unroll
    for (int j = 0; j < 2; j++)
#pragma unroll
      for (int i = 0; i < 8; i++) {
        int gr = rowBase + wrow + t * 16 + rlo + i;
        int gc = colBase + wcol + j * 16 + l15;
        float v = acc[t][j][i];
        if (ACT == 1) v = gelu_tanh(v);
        if (SCALE)    v *= rowscale[(size_t)gr * rsStride];
        if (ADD)      v += Cin[(size_t)gr * ldo + gc];
        if (OUTBF)    outB[(size_t)gr * ldo + gc] = (bf16)v;
        else          outF[(size_t)gr * ldo + gc] = v;
      }
}

// ---------------------------------------------------------------------------
// Flash attention: one block = (batch b, head h, 128-query tile).
// 8 waves x 16 queries. Q frags in registers; K staged row-major in LDS;
// V staged transposed; P goes through per-wave LDS for C->A layout transpose.
// Online softmax in f32; row reductions use DPP16 ROW_XMASK (VALU, no LDS).
// ---------------------------------------------------------------------------
__global__ __launch_bounds__(256) void attn_kernel(const bf16* __restrict__ qkv,
                                                   bf16* __restrict__ ctx) {
  const int qb   = blockIdx.x * 128;
  const int h    = blockIdx.y;
  const int b    = blockIdx.z;
  const int tid  = threadIdx.x;
  const int lane = tid & 31;
  const int wid  = tid >> 5;
  const int l15  = lane & 15;
  const int aoff = (lane & 16) ? 8 : 0;
  const int boff = (lane & 16) ? 8 : 0;
  const int ldq  = 3 * DMODEL;

  __shared__ alignas(16) bf16 Kt[32][72];      // 32 keys x 64 hd (+pad)
  __shared__ alignas(16) bf16 Vt[64][40];      // transposed: hd x 32 keys (+pad)
  __shared__ alignas(16) bf16 Pw[8][16][40];   // per-wave P tile 16x32 (+pad)

  // Q fragments: this wave's 16 queries, two 32-wide head-dim chunks
  const bf16* qrow = qkv + (size_t)(b * SEQ + qb + wid * 16 + l15) * ldq + h * HDIM;
  v16bf qf[2];
#pragma unroll
  for (int c = 0; c < 2; c++) {
    const bf16* p = qrow + c * 32 + aoff;
    qf[c] = cat16(*(const v8bf*)p, *(const v8bf*)(p + 16));
  }

  v8f o[4] = {};
  float mrun[8], lrun[8];
#pragma unroll
  for (int i = 0; i < 8; i++) { mrun[i] = -3.0e38f; lrun[i] = 0.f; }
  const int rlo  = (lane >> 4) << 3;
  const int rowB = qb + wid * 16 + rlo;

  for (int kb = 0; kb < qb + 128; kb += 32) {
    {  // stage K tile (32 keys x 64 hd), row-major
      int key = tid >> 3, c8 = (tid & 7) * 8;
      *(v8bf*)&Kt[key][c8] =
          *(const v8bf*)(qkv + (size_t)(b * SEQ + kb + key) * ldq + DMODEL + h * HDIM + c8);
    }
    {  // stage V^T (hd x keys): coalesced global read, scalar transpose into LDS
      int key = tid >> 3, d8 = (tid & 7) * 8;
      v8bf v = *(const v8bf*)(qkv + (size_t)(b * SEQ + kb + key) * ldq + 2 * DMODEL + h * HDIM + d8);
#pragma unroll
      for (int q = 0; q < 8; q++) Vt[d8 + q][key] = v[q];
    }
    __syncthreads();

    // S = Q K^T  (16 queries x 32 keys per wave)
    v8f s[2] = {};
#pragma unroll
    for (int c = 0; c < 2; c++)
#pragma unroll
      for (int half = 0; half < 2; half++) {
        const bf16* p = &Kt[half * 16 + l15][c * 32 + boff];
        v16bf kf = cat16(*(const v8bf*)p, *(const v8bf*)(p + 16));
        s[half] = WMMA_BF16(qf[c], kf, s[half]);
      }

    // online softmax (scale 1/sqrt(64) = 0.125, causal mask)
#pragma unroll
    for (int i = 0; i < 8; i++) {
      const int row = rowB + i;
      float s0 = s[0][i] * 0.125f;
      float s1 = s[1][i] * 0.125f;
      if (kb + l15 > row)      s0 = -3.0e38f;
      if (kb + 16 + l15 > row) s1 = -3.0e38f;
      float nm = fmaxf(mrun[i], rowmax16(fmaxf(s0, s1)));
      float p0 = __expf(s0 - nm), p1 = __expf(s1 - nm);
      float ps = rowsum16(p0 + p1);
      float sc = __expf(mrun[i] - nm);
      lrun[i] = lrun[i] * sc + ps;
      mrun[i] = nm;
#pragma unroll
      for (int j = 0; j < 4; j++) o[j][i] *= sc;
      Pw[wid][rlo + i][l15]      = (bf16)p0;   // C-layout -> LDS
      Pw[wid][rlo + i][16 + l15] = (bf16)p1;
    }

    // O += P V   (P re-read in A-operand layout from wave-private LDS)
    {
      const bf16* p = &Pw[wid][l15][aoff];
      v16bf pa = cat16(*(const v8bf*)p, *(const v8bf*)(p + 16));
#pragma unroll
      for (int j = 0; j < 4; j++) {
        const bf16* vp = &Vt[j * 16 + l15][boff];
        v16bf vb = cat16(*(const v8bf*)vp, *(const v8bf*)(vp + 16));
        o[j] = WMMA_BF16(pa, vb, o[j]);
      }
    }
    __syncthreads();
  }

  // normalize and write ctx (bf16, token-major)
#pragma unroll
  for (int i = 0; i < 8; i++) {
    float inv = 1.0f / lrun[i];
    bf16* cr = ctx + (size_t)(b * SEQ + rowB + i) * DMODEL + h * HDIM;
#pragma unroll
    for (int j = 0; j < 4; j++) cr[j * 16 + l15] = (bf16)(o[j][i] * inv);
  }
}

// ---------------------------------------------------------------------------
// LayerNorm (f32 in, bf16 out). One block per token, D = 1024.
// ---------------------------------------------------------------------------
__global__ __launch_bounds__(256) void ln_cast(const float* __restrict__ x,
                                               const float* __restrict__ g,
                                               const float* __restrict__ be,
                                               bf16* __restrict__ out) {
  const int row = blockIdx.x;
  const int tid = threadIdx.x;
  const float* xr = x + (size_t)row * DMODEL;
  float v[4], s = 0.f, s2 = 0.f;
#pragma unroll
  for (int i = 0; i < 4; i++) { v[i] = xr[tid + 256 * i]; s += v[i]; s2 += v[i] * v[i]; }
#pragma unroll
  for (int mk = 16; mk; mk >>= 1) { s += __shfl_xor(s, mk); s2 += __shfl_xor(s2, mk); }
  __shared__ float rs[8], rs2[8];
  if ((tid & 31) == 0) { rs[tid >> 5] = s; rs2[tid >> 5] = s2; }
  __syncthreads();
  s = 0.f; s2 = 0.f;
#pragma unroll
  for (int i = 0; i < 8; i++) { s += rs[i]; s2 += rs2[i]; }
  const float mu  = s * (1.0f / DMODEL);
  const float var = s2 * (1.0f / DMODEL) - mu * mu;
  const float r   = rsqrtf(var + 1e-5f);
#pragma unroll
  for (int i = 0; i < 4; i++) {
    int c = tid + 256 * i;
    out[(size_t)row * DMODEL + c] = (bf16)((v[i] - mu) * r * g[c] + be[c]);
  }
}

// ---------------------------------------------------------------------------
// Router: logits = m @ gate_w^T (E=8), top-2 softmax -> dense_w, atomic counts.
// One wave per token.
// ---------------------------------------------------------------------------
__global__ __launch_bounds__(256) void gate_router(const bf16* __restrict__ m,
                                                   const float* __restrict__ gw,
                                                   float* __restrict__ densew,
                                                   float* __restrict__ counts) {
  const int lane  = threadIdx.x & 31;
  const int token = blockIdx.x * 8 + (threadIdx.x >> 5);
  const bf16* mr  = m + (size_t)token * DMODEL;
  float acc[NEXP] = {};
  for (int d = lane; d < DMODEL; d += 32) {
    float mv = (float)mr[d];
#pragma unroll
    for (int e = 0; e < NEXP; e++) acc[e] += mv * gw[e * DMODEL + d];
  }
#pragma unroll
  for (int e = 0; e < NEXP; e++)
#pragma unroll
    for (int mk = 16; mk; mk >>= 1) acc[e] += __shfl_xor(acc[e], mk);
  if (lane == 0) {
    int i0 = 0; float v0 = acc[0];
#pragma unroll
    for (int e = 1; e < NEXP; e++) if (acc[e] > v0) { v0 = acc[e]; i0 = e; }
    int i1 = (i0 == 0) ? 1 : 0; float v1 = -3.0e38f;
#pragma unroll
    for (int e = 0; e < NEXP; e++) if (e != i0 && acc[e] > v1) { v1 = acc[e]; i1 = e; }
    float e1  = __expf(v1 - v0);        // v0 >= v1
    float inv = 1.0f / (1.0f + e1);
    float* dwr = densew + (size_t)token * NEXP;
#pragma unroll
    for (int e = 0; e < NEXP; e++) dwr[e] = 0.f;
    dwr[i0] = inv;
    dwr[i1] = e1 * inv;
    atomicAdd(&counts[i0], 1.0f);
    atomicAdd(&counts[i1], 1.0f);
  }
}

__global__ void zero_counts(float* c) { if (threadIdx.x < NEXP) c[threadIdx.x] = 0.f; }

__global__ void router_loss(const float* __restrict__ counts, float* __restrict__ out) {
  if (threadIdx.x == 0) {
    float s = 0.f;
    for (int e = 0; e < NEXP; e++) s += counts[e];
    float mean = s * (1.0f / NEXP);
    float var = 0.f;
    for (int e = 0; e < NEXP; e++) { float d = counts[e] - mean; var += d * d; }
    var *= (1.0f / (NEXP - 1));   // ddof=1
    out[0] = 0.01f * sqrtf(var) / (mean + 1e-6f);
  }
}

__global__ void copy_f32(const float* __restrict__ s, float* __restrict__ d, size_t n) {
  size_t i  = (size_t)blockIdx.x * blockDim.x + threadIdx.x;
  size_t st = (size_t)gridDim.x * blockDim.x;
  for (; i < n; i += st) d[i] = s[i];
}

// ---------------------------------------------------------------------------
// Host orchestration
// ---------------------------------------------------------------------------
extern "C" void kernel_launch(void* const* d_in, const int* in_sizes, int n_in,
                              void* d_out, int out_size, void* d_ws, size_t ws_size,
                              hipStream_t stream) {
  const float* x      = (const float*)d_in[0];
  const float* ln1_g  = (const float*)d_in[1];
  const float* ln1_b  = (const float*)d_in[2];
  const float* w_qkv  = (const float*)d_in[3];
  const float* w_proj = (const float*)d_in[4];
  const float* ln2_g  = (const float*)d_in[5];
  const float* ln2_b  = (const float*)d_in[6];
  const float* gate_w = (const float*)d_in[7];
  const float* fc1_w  = (const float*)d_in[8];
  const float* fc2_w  = (const float*)d_in[9];
  float* out = (float*)d_out;

  char* ws = (char*)d_ws;
  size_t off = 0;
  auto take = [&](size_t bytes) {
    char* p = ws + off;
    off = (off + bytes + 255) & ~(size_t)255;
    return p;
  };
  bf16*  hb     = (bf16*)  take((size_t)TOK * DMODEL * 2);       // LN1(x), bf16
  bf16*  qkvb   = (bf16*)  take((size_t)TOK * 3 * DMODEL * 2);   // qkv, bf16
  bf16*  ctxb   = (bf16*)  take((size_t)TOK * DMODEL * 2);       // attention out, bf16
  float* x1     = (float*) take((size_t)TOK * DMODEL * 4);       // x + proj, f32
  bf16*  mb     = (bf16*)  take((size_t)TOK * DMODEL * 2);       // LN2(x1), bf16
  float* densew = (float*) take((size_t)TOK * NEXP * 4);         // routing weights
  float* counts = (float*) take(256);
  bf16*  h1b    = (bf16*)  take((size_t)TOK * DFF * 2);          // expert hidden, bf16

  // 1) LN1 -> bf16
  ln_cast<<<TOK, 256, 0, stream>>>(x, ln1_g, ln1_b, hb);
  // 2) qkv = h @ w_qkv^T   (4096 x 3072 x 1024)
  gemm_xWT<0, false, false, true><<<dim3(3 * DMODEL / 128, TOK / 128), 256, 0, stream>>>(
      hb, DMODEL, w_qkv, DMODEL, nullptr, nullptr, qkvb, 3 * DMODEL, nullptr, 0, DMODEL);
  // 3) causal flash attention
  attn_kernel<<<dim3(SEQ / 128, NH, NBATCH), 256, 0, stream>>>(qkvb, ctxb);
  // 4) x1 = x + ctx @ w_proj^T
  gemm_xWT<0, true, false, false><<<dim3(DMODEL / 128, TOK / 128), 256, 0, stream>>>(
      ctxb, DMODEL, w_proj, DMODEL, x, x1, nullptr, DMODEL, nullptr, 0, DMODEL);
  // 5) LN2 -> bf16
  ln_cast<<<TOK, 256, 0, stream>>>(x1, ln2_g, ln2_b, mb);
  // 6) router
  zero_counts<<<1, 32, 0, stream>>>(counts);
  gate_router<<<TOK / 8, 256, 0, stream>>>(mb, gate_w, densew, counts);
  router_loss<<<1, 32, 0, stream>>>(counts, out + (size_t)TOK * DMODEL);
  // 7) y = x1
  copy_f32<<<1024, 256, 0, stream>>>(x1, out, (size_t)TOK * DMODEL);
  // 8) experts: y += dense_w[:,e] * (gelu(m@fc1^T) @ fc2^T)
  for (int e = 0; e < NEXP; e++) {
    gemm_xWT<1, false, false, true><<<dim3(DFF / 128, TOK / 128), 256, 0, stream>>>(
        mb, DMODEL, fc1_w + (size_t)e * DFF * DMODEL, DMODEL,
        nullptr, nullptr, h1b, DFF, nullptr, 0, DMODEL);
    gemm_xWT<0, true, true, false><<<dim3(DMODEL / 128, TOK / 128), 256, 0, stream>>>(
        h1b, DFF, fc2_w + (size_t)e * DMODEL * DFF, DFF,
        out, out, nullptr, DMODEL, densew + e, NEXP, DFF);
  }
  (void)in_sizes; (void)n_in; (void)out_size; (void)ws_size;
}
